// HetroGNNWithFusion_91207925498616
// MI455X (gfx1250) — matrix operand order repaired and checked by reading, not compiled, not run on hardware
//
#include <hip/hip_runtime.h>

// ---------------------------------------------------------------------------
// HetroGNNWithFusion on MI455X (gfx1250, wave32, WMMA bf16)
// Stage 1: h = tanh([rel|ent] @ Watt^T + b); att = h.u; softmax; agg = tanh(w.ent)
// Stage 2: y = relu(agg@Wl^T + origin@Wr^T) + origin; LayerNorm
// ---------------------------------------------------------------------------

#define DD 256   // feature dim
#define NN 30    // neighbors
#define KK 512   // 2*D (concat K)
#define CPAD 4   // concat row padding (ushorts) -> breaks 512-stride bank collisions

typedef __bf16 bf16_t;
typedef __attribute__((ext_vector_type(16))) bf16_t v16bf;
typedef __attribute__((ext_vector_type(8)))  float  v8f;

union Frag16 {
  v16bf v;
  unsigned short u[16];
  unsigned int   q[8];
  uint4          x[2];
};

__device__ __forceinline__ unsigned short f2bf(float f) {
  union { float f; unsigned int i; } c; c.f = f;
  unsigned int r = c.i + 0x7FFFu + ((c.i >> 16) & 1u);  // RNE
  return (unsigned short)(r >> 16);
}
__device__ __forceinline__ float bf2f(unsigned short u) {
  union { unsigned int i; float f; } c; c.i = ((unsigned int)u) << 16;
  return c.f;
}
__device__ __forceinline__ unsigned int pack2(float2 v) {
  return (unsigned int)f2bf(v.x) | ((unsigned int)f2bf(v.y) << 16);
}
// B-fragment-ordered LDS index for a [K x 256] bf16 weight matrix:
// tile (kt = k/32, nt = d/16), lane = (k/16 % 2)*16 + (d % 16), elem e = k % 16
__device__ __forceinline__ int fragIdx(int k, int d) {
  int kt = k >> 5, kh = (k >> 4) & 1, e = k & 15;
  int nt = d >> 4, dl = d & 15;
  return (((((kt << 4) | nt) << 5) | ((kh << 4) | dl)) << 4) | e;
}

// ------------------------------ Stage 1 ------------------------------------
__global__ __launch_bounds__(256) void gnn_stage1(
    const float* __restrict__ rel, const float* __restrict__ ent,
    const unsigned char* __restrict__ mask,
    const float* __restrict__ Watt, const float* __restrict__ batt,
    const float* __restrict__ uatt, const float* __restrict__ bu,
    float* __restrict__ agg, int Btot)
{
  __shared__ __align__(16) unsigned short sWtF[16 * 16 * 32 * 16]; // 256 KB, frag-ordered
  __shared__ __align__(16) unsigned short sC[32][KK + CPAD];       // concat bf16, rows padded
  __shared__ float sBias[DD];
  __shared__ float sU[DD];
  __shared__ float sAtt[32];
  __shared__ float sWgt[32];

  const int tid  = threadIdx.x;
  const int lane = tid & 31;
  const int wv   = tid >> 5;
  const int half = lane >> 4;
  const int lm   = lane & 15;

  // One-time per WG: Watt[d][k] (row-major 256x512) -> bf16 frag-ordered Wt[k][d]
  for (int i = tid; i < DD * KK; i += 256) {
    int d = i >> 9;
    int k = i & (KK - 1);
    sWtF[fragIdx(k, d)] = f2bf(Watt[i]);
  }
  for (int i = tid; i < DD; i += 256) { sBias[i] = batt[i]; sU[i] = uatt[i]; }
  const float bU = bu[0];
  __syncthreads();

  for (int b = blockIdx.x; b < Btot; b += gridDim.x) {
    const float* relb = rel + (size_t)b * NN * DD;
    const float* entb = ent + (size_t)b * NN * DD;
    // stream embeds once: f32 -> bf16 packed pairs into LDS concat [32][512]
    for (int i = tid; i < NN * DD / 2; i += 256) {
      int n  = i / (DD / 2);
      int d2 = i - n * (DD / 2);
      float2 vr = ((const float2*)relb)[i];
      float2 ve = ((const float2*)entb)[i];
      *(unsigned int*)&sC[n][2 * d2]      = pack2(vr);
      *(unsigned int*)&sC[n][DD + 2 * d2] = pack2(ve);
    }
    for (int i = tid; i < KK; i += 256) {              // zero pad rows 30,31
      *(unsigned int*)&sC[30 + (i >> 8)][2 * (i & 255)] = 0u;
    }
    if (tid < 32) sAtt[tid] = 0.0f;
    __syncthreads();

    // GEMM1: 2 Mtiles x 16 Ntiles; 8 waves * 4 tiles; K = 512 -> 16 WMMA/tile
    #pragma unroll
    for (int j = 0; j < 4; ++j) {
      int t  = (wv << 2) | j;
      int mt = t & 1, nt = t >> 1;
      int row  = mt * 16 + lm;        // A: lane holds row M = lane%16
      int dcol = (nt << 4) | lm;      // B/C: lane holds column N = lane%16
      v8f acc = {0.f, 0.f, 0.f, 0.f, 0.f, 0.f, 0.f, 0.f};
      const unsigned short* arow = &sC[row][0];
      #pragma unroll
      for (int kt = 0; kt < 16; ++kt) {
        Frag16 a, w;
        int k0 = (kt << 5) + (half << 4);   // A K layout: half0 {0..15}, half1 {16..31}
        #pragma unroll
        for (int e = 0; e < 4; ++e) {
          a.q[e]     = *(const unsigned int*)&arow[k0 + 2 * e];
          a.q[4 + e] = *(const unsigned int*)&arow[k0 + 8 + 2 * e];
        }
        const uint4* wp = (const uint4*)&sWtF[(((kt << 4) | nt) << 9) + (lane << 4)];
        w.x[0] = wp[0];
        w.x[1] = wp[1];
        acc = __builtin_amdgcn_wmma_f32_16x16x32_bf16(false, a.v, false, w.v,
                                                      (short)0, acc, false, false);
      }
      // h = tanh(acc + b[d]); att[m] += h*u[d], reduced over the 16 d-lanes in-register
      float uu = sU[dcol], bb = sBias[dcol];
      #pragma unroll
      for (int r = 0; r < 8; ++r) {
        float h = tanhf(acc[r] + bb);
        float p = h * uu;
        p += __shfl_xor(p, 1, 32);
        p += __shfl_xor(p, 2, 32);
        p += __shfl_xor(p, 4, 32);
        p += __shfl_xor(p, 8, 32);
        if (lm == 0) atomicAdd(&sAtt[mt * 16 + r + (half << 3)], p);
      }
    }
    __syncthreads();

    // masked softmax over n (wave 0)
    if (wv == 0) {
      bool valid = (lane < NN) && (mask[(size_t)b * NN + lane] == 0);
      float a = valid ? (sAtt[lane] + bU) : -__builtin_inff();
      float mx = a;
      mx = fmaxf(mx, __shfl_xor(mx, 16, 32));
      mx = fmaxf(mx, __shfl_xor(mx, 8, 32));
      mx = fmaxf(mx, __shfl_xor(mx, 4, 32));
      mx = fmaxf(mx, __shfl_xor(mx, 2, 32));
      mx = fmaxf(mx, __shfl_xor(mx, 1, 32));
      float e = valid ? __expf(a - mx) : 0.0f;
      float s = e;
      s += __shfl_xor(s, 16, 32);
      s += __shfl_xor(s, 8, 32);
      s += __shfl_xor(s, 4, 32);
      s += __shfl_xor(s, 2, 32);
      s += __shfl_xor(s, 1, 32);
      sWgt[lane] = e / s;
    }
    __syncthreads();

    // agg[d] = tanh(sum_n w[n] * ent[n][d]) ; ent bf16 already in LDS
    {
      int d = tid;  // 256 threads == DD
      float s = 0.0f;
      #pragma unroll
      for (int n = 0; n < NN; ++n) s += sWgt[n] * bf2f(sC[n][DD + d]);
      agg[(size_t)b * DD + d] = tanhf(s);
    }
    __syncthreads();
  }
}

// ------------------------------ Stage 2 ------------------------------------
__global__ __launch_bounds__(256) void gnn_stage2(
    const float* __restrict__ agg, const float* __restrict__ origin,
    const float* __restrict__ Wl, const float* __restrict__ Wr,
    const float* __restrict__ gamma, const float* __restrict__ beta,
    float* __restrict__ out, int Btot)
{
  __shared__ __align__(16) unsigned short sWF[16 * 16 * 32 * 16]; // [Wl;Wr]^T frag-ordered
  __shared__ __align__(16) unsigned short sA[32][KK + CPAD];      // [agg|origin] bf16
  __shared__ float sSum[32];
  __shared__ float sSq[32];

  const int tid  = threadIdx.x;
  const int lane = tid & 31;
  const int wv   = tid >> 5;
  const int half = lane >> 4;
  const int lm   = lane & 15;

  // Wcat[k][i] = (k<256) ? Wl[i][k] : Wr[i][k-256]
  for (int i = tid; i < DD * DD; i += 256) {
    int ii = i >> 8, k = i & 255;
    sWF[fragIdx(k, ii)]       = f2bf(Wl[i]);
    sWF[fragIdx(k + 256, ii)] = f2bf(Wr[i]);
  }
  __syncthreads();

  const int nblk = Btot >> 5;  // 32 batch rows per block-tile
  for (int blk = blockIdx.x; blk < nblk; blk += gridDim.x) {
    int b0 = blk << 5;
    for (int i = tid; i < 32 * DD / 2; i += 256) {
      int m = i >> 7, d2 = i & 127;
      float2 va = ((const float2*)(agg    + (size_t)(b0 + m) * DD))[d2];
      float2 vo = ((const float2*)(origin + (size_t)(b0 + m) * DD))[d2];
      *(unsigned int*)&sA[m][2 * d2]      = pack2(va);
      *(unsigned int*)&sA[m][DD + 2 * d2] = pack2(vo);
    }
    if (tid < 32) { sSum[tid] = 0.0f; sSq[tid] = 0.0f; }
    __syncthreads();

    float yv[4][8];
    #pragma unroll
    for (int j = 0; j < 4; ++j) {
      int t  = (wv << 2) | j;
      int mt = t & 1, nt = t >> 1;
      int row  = mt * 16 + lm;
      int dcol = (nt << 4) | lm;
      v8f acc = {0.f, 0.f, 0.f, 0.f, 0.f, 0.f, 0.f, 0.f};
      const unsigned short* arow = &sA[row][0];
      #pragma unroll
      for (int kt = 0; kt < 16; ++kt) {
        Frag16 a, w;
        int k0 = (kt << 5) + (half << 4);
        #pragma unroll
        for (int e = 0; e < 4; ++e) {
          a.q[e]     = *(const unsigned int*)&arow[k0 + 2 * e];
          a.q[4 + e] = *(const unsigned int*)&arow[k0 + 8 + 2 * e];
        }
        const uint4* wp = (const uint4*)&sWF[(((kt << 4) | nt) << 9) + (lane << 4)];
        w.x[0] = wp[0];
        w.x[1] = wp[1];
        acc = __builtin_amdgcn_wmma_f32_16x16x32_bf16(false, a.v, false, w.v,
                                                      (short)0, acc, false, false);
      }
      // y = relu(x1) + origin (exact f32 origin re-read); accumulate LN stats
      #pragma unroll
      for (int r = 0; r < 8; ++r) {
        int m = mt * 16 + r + (half << 3);
        float x  = fmaxf(acc[r], 0.0f);
        float yy = x + origin[(size_t)(b0 + m) * DD + dcol];
        yv[j][r] = yy;
        float p = yy, q = yy * yy;
        p += __shfl_xor(p, 1, 32); q += __shfl_xor(q, 1, 32);
        p += __shfl_xor(p, 2, 32); q += __shfl_xor(q, 2, 32);
        p += __shfl_xor(p, 4, 32); q += __shfl_xor(q, 4, 32);
        p += __shfl_xor(p, 8, 32); q += __shfl_xor(q, 8, 32);
        if (lm == 0) { atomicAdd(&sSum[m], p); atomicAdd(&sSq[m], q); }
      }
    }
    __syncthreads();

    #pragma unroll
    for (int j = 0; j < 4; ++j) {
      int t  = (wv << 2) | j;
      int mt = t & 1, nt = t >> 1;
      int dcol = (nt << 4) | lm;
      float g = gamma[dcol], bb = beta[dcol];
      #pragma unroll
      for (int r = 0; r < 8; ++r) {
        int m = mt * 16 + r + (half << 3);
        float mu  = sSum[m] * (1.0f / DD);
        float var = sSq[m] * (1.0f / DD) - mu * mu;
        out[(size_t)(b0 + m) * DD + dcol] =
            (yv[j][r] - mu) * rsqrtf(var + 1e-5f) * g + bb;
      }
    }
    __syncthreads();
  }
}

// ------------------------------ Launch -------------------------------------
extern "C" void kernel_launch(void* const* d_in, const int* in_sizes, int n_in,
                              void* d_out, int out_size, void* d_ws, size_t ws_size,
                              hipStream_t stream) {
  const float* rel    = (const float*)d_in[0];
  const float* ent    = (const float*)d_in[1];
  const float* origin = (const float*)d_in[2];
  const unsigned char* mask = (const unsigned char*)d_in[3];
  const float* Watt   = (const float*)d_in[4];
  const float* batt   = (const float*)d_in[5];
  const float* uatt   = (const float*)d_in[6];
  const float* bu     = (const float*)d_in[7];
  const float* Wl     = (const float*)d_in[8];
  const float* Wr     = (const float*)d_in[9];
  const float* gamma  = (const float*)d_in[10];
  const float* beta   = (const float*)d_in[11];

  const int Btot = in_sizes[2] / DD;          // origin is [B, 256]
  float* aggBuf  = (float*)d_ws;              // [B, 256] f32 intermediate (16 MB)

  gnn_stage1<<<512, 256, 0, stream>>>(rel, ent, mask, Watt, batt, uatt, bu,
                                      aggBuf, Btot);
  gnn_stage2<<<128, 256, 0, stream>>>(aggBuf, origin, Wl, Wr, gamma, beta,
                                      (float*)d_out, Btot);
}
